// DeepSetsEncoder_41489384079480
// MI455X (gfx1250) — matrix operand order repaired and checked by reading, not compiled, not run on hardware
//
#include <hip/hip_runtime.h>

// ---------------------------------------------------------------------------
// DeepSets encoder for MI455X (gfx1250, wave32, WMMA).
// phi MLP: 16 -> 256 (LN,ReLU) -> 256 (LN,ReLU) -> 128, ragged mean pool,
// broadcast back. bf16 WMMA operands, fp32 accumulation.
// ---------------------------------------------------------------------------

typedef __attribute__((ext_vector_type(8)))  float          v8f;
typedef __attribute__((ext_vector_type(16))) __bf16         v16bf;

union Frag16 {
  v16bf b;
  uint4 q[2];
};

#define EPS      1e-5f
#define D_IN     16
#define HDIM     256
#define D_OUT    128
#define NT_H     16                 // 256/16 N-tiles for hidden dim
#define NT_O     8                  // 128/16 N-tiles for output dim
#define KC_H     8                  // 256/32 K-chunks
#define NWAVES   8

#define W1S_ELEMS (16 * 32 * 16)        // 8192  bf16 (K padded to 32)
#define W2S_ELEMS (8 * 16 * 32 * 16)    // 65536 bf16
#define W3S_ELEMS (8 * 8 * 32 * 16)     // 32768 bf16
#define HS_ELEMS_PER_WAVE (16 * 256)    // 4096  bf16

__device__ __forceinline__ int seg_search(const int* __restrict__ offsets, int B, int p) {
  int lo = 0, hi = B - 1;
  while (lo < hi) {
    int mid = (lo + hi + 1) >> 1;
    if (offsets[mid] <= p) lo = mid; else hi = mid - 1;
  }
  return lo;
}

// ---------------------------------------------------------------------------
// Kernel 1: exclusive prefix-sum of counts (block 0) + zero the sums table.
// Assumes B <= 2048 (B = 2000 here). blockDim must be 256.
// ---------------------------------------------------------------------------
__global__ void setup_offsets_zero(const int* __restrict__ counts,
                                   int* __restrict__ offsets,
                                   float* __restrict__ sums,
                                   int B, int sumsElems) {
  __shared__ int partial[257];
  if (blockIdx.x == 0) {
    int t = threadIdx.x;
    int base = t * 8;
    int loc[8];
    int s = 0;
#pragma unroll
    for (int i = 0; i < 8; ++i) {
      int idx = base + i;
      int c = (idx < B) ? counts[idx] : 0;
      loc[i] = s;
      s += c;
    }
    partial[t] = s;
    __syncthreads();
    if (t == 0) {
      int acc = 0;
      for (int i = 0; i < 256; ++i) { int tmp = partial[i]; partial[i] = acc; acc += tmp; }
      partial[256] = acc;
    }
    __syncthreads();
#pragma unroll
    for (int i = 0; i < 8; ++i) {
      int idx = base + i;
      if (idx < B) offsets[idx] = partial[t] + loc[i];
    }
    if (t == 0) offsets[B] = partial[256];
  }
  int gid = blockIdx.x * blockDim.x + threadIdx.x;
  if (gid < sumsElems) sums[gid] = 0.f;
}

// ---------------------------------------------------------------------------
// Kernel 2: per-point MLP via WMMA + fused ragged segment-sum (atomics).
// One wave handles one 16-point tile per iteration; all weights live in LDS
// pre-swizzled into WMMA B-fragment order (32 contiguous bytes per lane).
// ---------------------------------------------------------------------------
__global__ void __launch_bounds__(256)
deepsets_phi_pool(const float* __restrict__ z,
                  const float* __restrict__ W1, const float* __restrict__ b1,
                  const float* __restrict__ g1, const float* __restrict__ be1,
                  const float* __restrict__ W2, const float* __restrict__ b2,
                  const float* __restrict__ g2, const float* __restrict__ be2,
                  const float* __restrict__ W3, const float* __restrict__ b3,
                  const int* __restrict__ offsets, float* __restrict__ sums,
                  int B, int numTiles) {
  extern __shared__ char smem[];
  __bf16* w1s   = (__bf16*)smem;
  __bf16* w2s   = w1s + W1S_ELEMS;
  __bf16* w3s   = w2s + W2S_ELEMS;
  __bf16* hsAll = w3s + W3S_ELEMS;
  float* b1s  = (float*)(hsAll + HS_ELEMS_PER_WAVE * NWAVES);
  float* g1s  = b1s + 256;
  float* be1s = g1s + 256;
  float* b2s  = be1s + 256;
  float* g2s  = b2s + 256;
  float* be2s = g2s + 256;
  float* b3s  = be2s + 256;

  const int tid = threadIdx.x;

  // ---- Stage weights into LDS in B-fragment-major order (bf16) ----
  // Fragment layout (16-bit B, 32x16): lane l holds col N = 16j+(l&15),
  // K = 32k + 16*(e>>3) + (l<16 ? 0 : 8) + (e&7) for e = 0..15, contiguous per lane.
  for (int idx = tid; idx < W1S_ELEMS; idx += 256) {
    int j  = idx >> 9;
    int r  = idx & 511;
    int ln = r >> 4;
    int e  = r & 15;
    int K  = ((e >> 3) << 4) + ((ln >> 4) << 3) + (e & 7);          // layer-1 K padded 16 -> 32
    int n  = (j << 4) + (ln & 15);
    float v = (K < D_IN) ? W1[K * HDIM + n] : 0.f;
    w1s[idx] = (__bf16)v;
  }
  for (int idx = tid; idx < W2S_ELEMS; idx += 256) {
    int k  = idx >> 13;
    int r  = idx & 8191;
    int j  = r >> 9;
    int r2 = r & 511;
    int ln = r2 >> 4;
    int e  = r2 & 15;
    int K  = (k << 5) + ((e >> 3) << 4) + ((ln >> 4) << 3) + (e & 7);
    int n  = (j << 4) + (ln & 15);
    w2s[idx] = (__bf16)W2[K * HDIM + n];
  }
  for (int idx = tid; idx < W3S_ELEMS; idx += 256) {
    int k  = idx >> 12;
    int r  = idx & 4095;
    int j  = r >> 9;
    int r2 = r & 511;
    int ln = r2 >> 4;
    int e  = r2 & 15;
    int K  = (k << 5) + ((e >> 3) << 4) + ((ln >> 4) << 3) + (e & 7);
    int n  = (j << 4) + (ln & 15);
    w3s[idx] = (__bf16)W3[K * D_OUT + n];
  }
  {
    int i = tid;
    b1s[i] = b1[i]; g1s[i] = g1[i]; be1s[i] = be1[i];
    b2s[i] = b2[i]; g2s[i] = g2[i]; be2s[i] = be2[i];
    if (i < D_OUT) b3s[i] = b3[i];
  }
  __syncthreads();

  const int lane   = tid & 31;
  const int wave   = tid >> 5;
  const int halfHi = lane >> 4;          // 0: lanes 0-15, 1: lanes 16-31
  const int col    = lane & 15;
  __bf16* hs = hsAll + (wave << 12);     // 16x256 bf16 wave scratch

  const int waveId = (blockIdx.x << 3) + wave;
  const int nWaves = gridDim.x << 3;
  const v8f vzero = {0.f, 0.f, 0.f, 0.f, 0.f, 0.f, 0.f, 0.f};

  for (int tile = waveId; tile < numTiles; tile += nWaves) {
    const int p0 = tile << 4;
    if (tile + nWaves < numTiles)
      __builtin_prefetch(z + (size_t)(tile + nWaves) * (16 * D_IN), 0, 0);

    // Segment id for tile row 'col' (lane l searches for row l&15).
    int segc     = seg_search(offsets, B, p0 + col);
    int segFirst = __shfl(segc, 0, 32);
    int segLast  = __shfl(segc, 15, 32);

    // ---------------- Layer 1: [16,16(pad 32)] x [32,256] ----------------
    // A-frag (16-bit 16x32): lane l, M = l&15; lanes 0-15 hold K 0..7,16..23,
    // lanes 16-31 hold K 8..15,24..31. K >= 16 is zero padding.
    Frag16 a1;
    {
      const float* zr = z + (size_t)(p0 + col) * D_IN + (halfHi << 3);
      const float4 f0 = ((const float4*)zr)[0];
      const float4 f1 = ((const float4*)zr)[1];
      v16bf u;
      u[0] = (__bf16)f0.x; u[1] = (__bf16)f0.y;
      u[2] = (__bf16)f0.z; u[3] = (__bf16)f0.w;
      u[4] = (__bf16)f1.x; u[5] = (__bf16)f1.y;
      u[6] = (__bf16)f1.z; u[7] = (__bf16)f1.w;
      u[8]  = (__bf16)0.f; u[9]  = (__bf16)0.f;
      u[10] = (__bf16)0.f; u[11] = (__bf16)0.f;
      u[12] = (__bf16)0.f; u[13] = (__bf16)0.f;
      u[14] = (__bf16)0.f; u[15] = (__bf16)0.f;
      a1.b = u;
    }
    v8f c[NT_H];
#pragma unroll
    for (int j = 0; j < NT_H; ++j) {
      Frag16 bf;
      const uint4* wp = (const uint4*)(w1s + (((j << 5) + lane) << 4));
      bf.q[0] = wp[0]; bf.q[1] = wp[1];
      c[j] = __builtin_amdgcn_wmma_f32_16x16x32_bf16(false, a1.b, false, bf.b,
                                                     (short)0, vzero, false, false);
    }

    // ---- bias + LayerNorm + ReLU -> hs (bf16 row-major 16x256) ----
    // C layout: lane holds rows M = v + 8*halfHi, col N = 16j + (l&15).
#pragma unroll
    for (int v = 0; v < 8; ++v) {
      float s = 0.f, sq = 0.f;
#pragma unroll
      for (int j = 0; j < NT_H; ++j) {
        float val = c[j][v] + b1s[(j << 4) + col];
        c[j][v] = val;
        s += val; sq += val * val;
      }
#pragma unroll
      for (int m = 1; m < 16; m <<= 1) {
        s  += __shfl_xor(s,  m, 32);
        sq += __shfl_xor(sq, m, 32);
      }
      float mean = s * (1.f / 256.f);
      float var  = sq * (1.f / 256.f) - mean * mean;
      float rstd = rsqrtf(var + EPS);
      int row = v + (halfHi << 3);
#pragma unroll
      for (int j = 0; j < NT_H; ++j) {
        int n = (j << 4) + col;
        float val = (c[j][v] - mean) * rstd * g1s[n] + be1s[n];
        val = fmaxf(val, 0.f);
        hs[(row << 8) + n] = (__bf16)val;
      }
    }

    // ---------------- Layer 2: [16,256] x [256,256] ----------------
#pragma unroll
    for (int j = 0; j < NT_H; ++j) c[j] = vzero;
#pragma unroll
    for (int k = 0; k < KC_H; ++k) {
      Frag16 a;
      const uint4* hp = (const uint4*)(hs + (col << 8) + (k << 5) + (halfHi << 3));
      a.q[0] = hp[0];          // K  base .. base+7
      a.q[1] = hp[2];          // K  base+16 .. base+23
#pragma unroll
      for (int j = 0; j < NT_H; ++j) {
        Frag16 bf;
        const uint4* wp = (const uint4*)(w2s + ((((k << 4) + j) * 32 + lane) << 4));
        bf.q[0] = wp[0]; bf.q[1] = wp[1];
        c[j] = __builtin_amdgcn_wmma_f32_16x16x32_bf16(false, a.b, false, bf.b,
                                                       (short)0, c[j], false, false);
      }
    }
#pragma unroll
    for (int v = 0; v < 8; ++v) {
      float s = 0.f, sq = 0.f;
#pragma unroll
      for (int j = 0; j < NT_H; ++j) {
        float val = c[j][v] + b2s[(j << 4) + col];
        c[j][v] = val;
        s += val; sq += val * val;
      }
#pragma unroll
      for (int m = 1; m < 16; m <<= 1) {
        s  += __shfl_xor(s,  m, 32);
        sq += __shfl_xor(sq, m, 32);
      }
      float mean = s * (1.f / 256.f);
      float var  = sq * (1.f / 256.f) - mean * mean;
      float rstd = rsqrtf(var + EPS);
      int row = v + (halfHi << 3);
#pragma unroll
      for (int j = 0; j < NT_H; ++j) {
        int n = (j << 4) + col;
        float val = (c[j][v] - mean) * rstd * g2s[n] + be2s[n];
        val = fmaxf(val, 0.f);
        hs[(row << 8) + n] = (__bf16)val;
      }
    }

    // ---------------- Layer 3: [16,256] x [256,128] ----------------
    v8f c3[NT_O];
#pragma unroll
    for (int j = 0; j < NT_O; ++j) c3[j] = vzero;
#pragma unroll
    for (int k = 0; k < KC_H; ++k) {
      Frag16 a;
      const uint4* hp = (const uint4*)(hs + (col << 8) + (k << 5) + (halfHi << 3));
      a.q[0] = hp[0]; a.q[1] = hp[2];
#pragma unroll
      for (int j = 0; j < NT_O; ++j) {
        Frag16 bf;
        const uint4* wp = (const uint4*)(w3s + ((((k << 3) + j) * 32 + lane) << 4));
        bf.q[0] = wp[0]; bf.q[1] = wp[1];
        c3[j] = __builtin_amdgcn_wmma_f32_16x16x32_bf16(false, a.b, false, bf.b,
                                                        (short)0, c3[j], false, false);
      }
    }
#pragma unroll
    for (int j = 0; j < NT_O; ++j) {
#pragma unroll
      for (int v = 0; v < 8; ++v) c3[j][v] += b3s[(j << 4) + col];
    }

    // ---- Fused ragged segment-sum accumulation ----
    if (segFirst == segLast) {
      // Whole 16-row tile in one segment: reduce rows in registers, then one
      // cross-half shuffle, 8 atomics per half-wave.
#pragma unroll
      for (int j = 0; j < NT_O; ++j) {
        float partial = 0.f;
#pragma unroll
        for (int v = 0; v < 8; ++v) partial += c3[j][v];
        partial += __shfl_xor(partial, 16, 32);
        if (lane < 16)
          atomicAdd(&sums[(size_t)segFirst * D_OUT + (j << 4) + col], partial);
      }
    } else {
      // Tile crosses a segment boundary (~3% of tiles): per-row atomics.
#pragma unroll
      for (int v = 0; v < 8; ++v) {
        int row  = v + (halfHi << 3);
        int segr = __shfl(segc, row, 32);
#pragma unroll
        for (int j = 0; j < NT_O; ++j)
          atomicAdd(&sums[(size_t)segr * D_OUT + (j << 4) + col], c3[j][v]);
      }
    }
  }
}

// ---------------------------------------------------------------------------
// Kernel 3: out[p] = sums[seg(p)] / count(seg(p)), streamed with float4.
// One thread emits a 32-float chunk (128 B) per point-quadrant.
// ---------------------------------------------------------------------------
__global__ void gather_bcast(const int* __restrict__ offsets,
                             const float* __restrict__ sums,
                             float* __restrict__ out, int B, int N) {
  int idx = blockIdx.x * blockDim.x + threadIdx.x;
  int total = N * 4;
  if (idx >= total) return;
  int p  = idx >> 2;
  int c0 = (idx & 3) << 5;
  int seg = seg_search(offsets, B, p);
  float inv = 1.f / (float)(offsets[seg + 1] - offsets[seg]);
  const float4* s4 = (const float4*)(sums + (size_t)seg * D_OUT + c0);
  float4* o4 = (float4*)(out + (size_t)p * D_OUT + c0);
#pragma unroll
  for (int i = 0; i < 8; ++i) {
    float4 v = s4[i];
    v.x *= inv; v.y *= inv; v.z *= inv; v.w *= inv;
    o4[i] = v;
  }
}

// ---------------------------------------------------------------------------
extern "C" void kernel_launch(void* const* d_in, const int* in_sizes, int n_in,
                              void* d_out, int out_size, void* d_ws, size_t ws_size,
                              hipStream_t stream) {
  const float* z   = (const float*)d_in[0];
  const float* W1  = (const float*)d_in[1];
  const float* b1  = (const float*)d_in[2];
  const float* g1  = (const float*)d_in[3];
  const float* be1 = (const float*)d_in[4];
  const float* W2  = (const float*)d_in[5];
  const float* b2  = (const float*)d_in[6];
  const float* g2  = (const float*)d_in[7];
  const float* be2 = (const float*)d_in[8];
  const float* W3  = (const float*)d_in[9];
  const float* b3  = (const float*)d_in[10];
  const int* counts = (const int*)d_in[11];

  const int N = in_sizes[0] / D_IN;      // 1,000,000
  const int B = in_sizes[11];            // 2,000
  float* out = (float*)d_out;

  // Workspace: [offsets: (B+1) ints][pad to 256B][sums: B*128 floats] ~ 1.03 MB
  int* offsets = (int*)d_ws;
  size_t sumsOff = (((size_t)(B + 1) * sizeof(int)) + 255) & ~(size_t)255;
  float* sums = (float*)((char*)d_ws + sumsOff);

  const int sumsElems = B * D_OUT;
  setup_offsets_zero<<<(sumsElems + 255) / 256, 256, 0, stream>>>(
      counts, offsets, sums, B, sumsElems);

  const int numTiles = (N + 15) / 16;    // 62,500 (N % 16 == 0)
  const size_t smemBytes =
      (size_t)(W1S_ELEMS + W2S_ELEMS + W3S_ELEMS + HS_ELEMS_PER_WAVE * NWAVES) * 2
      + (size_t)(6 * HDIM + D_OUT) * sizeof(float);   // ~279 KB of 320 KB WGP LDS
  deepsets_phi_pool<<<640, 256, smemBytes, stream>>>(
      z, W1, b1, g1, be1, W2, b2, g2, be2, W3, b3, offsets, sums, B, numTiles);

  const int total = N * 4;
  gather_bcast<<<(total + 255) / 256, 256, 0, stream>>>(offsets, sums, out, B, N);
}
